// TransformerBlock_89842125897796
// MI455X (gfx1250) — compile-verified, matrix operand order checked by reading, not run on hardware
//
#include <hip/hip_runtime.h>
#include <hip/hip_bf16.h>
#include <math.h>

typedef __attribute__((ext_vector_type(16))) __bf16 v16bf;
typedef __attribute__((ext_vector_type(8)))  float  v8f;
typedef int v4i_t __attribute__((vector_size(16)));
typedef unsigned short us;

#define DEV static __device__ __forceinline__
#define AS1 __attribute__((address_space(1)))
#define AS3 __attribute__((address_space(3)))

#if defined(__has_builtin)
#if __has_builtin(__builtin_amdgcn_global_load_async_to_lds_b128)
#define HAVE_ASYNC_BUILTIN 1
#endif
#if __has_builtin(__builtin_amdgcn_s_wait_asynccnt)
#define HAVE_WAIT_ASYNC_BUILTIN 1
#endif
#endif
#ifndef HAVE_ASYNC_BUILTIN
#define HAVE_ASYNC_BUILTIN 0
#endif
#ifndef HAVE_WAIT_ASYNC_BUILTIN
#define HAVE_WAIT_ASYNC_BUILTIN 0
#endif

union FragB16 { uint4 q[2]; v16bf v; };
union U4S8    { uint4 q; us s[8]; };

DEV us f2bf(float f) {
  unsigned u = __float_as_uint(f);
  u += 0x7FFFu + ((u >> 16) & 1u);          // round-to-nearest-even
  return (us)(u >> 16);
}
DEV v8f zero8() { v8f z = {0.f,0.f,0.f,0.f,0.f,0.f,0.f,0.f}; return z; }

DEV v8f wmma_bf16(const FragB16& a, const FragB16& b, v8f c) {
  return __builtin_amdgcn_wmma_f32_16x16x32_bf16(false, a.v, false, b.v,
                                                 (short)0, c, false, false);
}

// 16-byte async global -> LDS copy (per-lane addresses), ASYNCcnt-tracked.
DEV void async_copy16(us* lds, const us* g) {
#if HAVE_ASYNC_BUILTIN
  __builtin_amdgcn_global_load_async_to_lds_b128((AS1 v4i_t*)(us*)g,
                                                 (AS3 v4i_t*)lds, 0, 0);
#else
  unsigned loff = (unsigned)(unsigned long long)(AS3 us*)lds;
  unsigned long long ga = (unsigned long long)g;
  asm volatile("global_load_async_to_lds_b128 %0, %1, off"
               :: "v"(loff), "v"(ga) : "memory");
#endif
}
DEV void wait_async0() {
#if HAVE_WAIT_ASYNC_BUILTIN
  __builtin_amdgcn_s_wait_asynccnt(0);
#else
  asm volatile("s_wait_asynccnt 0" ::: "memory");
#endif
}

// max across each 16-lane group, pure VALU via DPP16 (no LDS round trips)
DEV float dppmax16(float v) {
  float t;
  t = __int_as_float(__builtin_amdgcn_update_dpp(0, __float_as_int(v), 0x0B1, 0xF, 0xF, true));
  v = fmaxf(v, t);   // quad_perm [1,0,3,2]
  t = __int_as_float(__builtin_amdgcn_update_dpp(0, __float_as_int(v), 0x04E, 0xF, 0xF, true));
  v = fmaxf(v, t);   // quad_perm [2,3,0,1]
  t = __int_as_float(__builtin_amdgcn_update_dpp(0, __float_as_int(v), 0x141, 0xF, 0xF, true));
  v = fmaxf(v, t);   // row_half_mirror
  t = __int_as_float(__builtin_amdgcn_update_dpp(0, __float_as_int(v), 0x140, 0xF, 0xF, true));
  v = fmaxf(v, t);   // row_mirror
  return v;
}

// ---------------- fp32 -> bf16 (row-major copy: context) --------------------
__global__ void k_cvt_bf16(const float* __restrict__ src, us* __restrict__ dst,
                           int n4) {
  int i = blockIdx.x * blockDim.x + threadIdx.x;
  if (i >= n4) return;
  float4 f = ((const float4*)src)[i];
  int o = i * 4;
  dst[o + 0] = f2bf(f.x); dst[o + 1] = f2bf(f.y);
  dst[o + 2] = f2bf(f.z); dst[o + 3] = f2bf(f.w);
}

// -------- fp32 W[K,N] -> bf16 Wt[N,K] (convert + transpose, 32x32 tiles) ----
__global__ __launch_bounds__(256)
void k_cvt_t(const float* __restrict__ src, us* __restrict__ dst, int K, int N) {
  __shared__ __align__(16) float t[32][36];
  const int kb = blockIdx.y * 32, nb = blockIdx.x * 32;
  {
    int r = threadIdx.x >> 3, c4 = (threadIdx.x & 7) << 2;
    float4 f = *(const float4*)(src + (size_t)(kb + r) * N + nb + c4);
    t[r][c4 + 0] = f.x; t[r][c4 + 1] = f.y; t[r][c4 + 2] = f.z; t[r][c4 + 3] = f.w;
  }
  __syncthreads();
  {
    int n = threadIdx.x >> 3, k4 = (threadIdx.x & 7) << 2;
    unsigned a0 = (unsigned)f2bf(t[k4 + 0][n]) | ((unsigned)f2bf(t[k4 + 1][n]) << 16);
    unsigned a1 = (unsigned)f2bf(t[k4 + 2][n]) | ((unsigned)f2bf(t[k4 + 3][n]) << 16);
    uint2 u; u.x = a0; u.y = a1;
    *(uint2*)(dst + (size_t)(nb + n) * K + kb + k4) = u;
  }
}

// -------- bf16 V[t, d] (strided heads) -> Vt[(b*H+h)*64 + d][T] -------------
__global__ __launch_bounds__(256)
void k_vt(const us* __restrict__ Vsrc, int rs, int col0,
          us* __restrict__ Vt, int T, int H) {
  __shared__ __align__(16) us tl[32][72];
  const int bh = blockIdx.y, b = bh / H, h = bh % H;
  const int t0 = blockIdx.x * 32;
  {
    int t = threadIdx.x >> 3, d8 = (threadIdx.x & 7) << 3;
    *(uint4*)&tl[t][d8] =
        *(const uint4*)(Vsrc + ((size_t)b * T + t0 + t) * rs + col0 + h * 64 + d8);
  }
  __syncthreads();
  {
    int d = threadIdx.x >> 2, t8 = (threadIdx.x & 3) << 3;
    U4S8 u;
    #pragma unroll
    for (int e = 0; e < 8; ++e) u.s[e] = tl[t8 + e][d];
    *(uint4*)(Vt + ((size_t)bh * 64 + d) * T + t0 + t8) = u.q;
  }
}

// ---------------- LayerNorm: one block per row (C = 1024) -------------------
__global__ __launch_bounds__(256)
void k_layernorm(const float* __restrict__ x, const float* __restrict__ g,
                 const float* __restrict__ b, us* __restrict__ out, int C) {
  const int row = blockIdx.x, tid = threadIdx.x;
  const float* xr = x + (size_t)row * C;
  float v[4], s = 0.f, s2 = 0.f;
  #pragma unroll
  for (int i = 0; i < 4; ++i) {
    float t = xr[tid + i * 256];
    v[i] = t; s += t; s2 += t * t;
  }
  #pragma unroll
  for (int off = 16; off; off >>= 1) {
    s  += __shfl_xor(s,  off, 32);
    s2 += __shfl_xor(s2, off, 32);
  }
  __shared__ float sh[16];
  int wid = tid >> 5, lane = tid & 31;
  if (lane == 0) { sh[wid] = s; sh[8 + wid] = s2; }
  __syncthreads();
  float ts = 0.f, ts2 = 0.f;
  #pragma unroll
  for (int i = 0; i < 8; ++i) { ts += sh[i]; ts2 += sh[8 + i]; }
  float mean = ts / C;
  float rstd = rsqrtf(ts2 / C - mean * mean + 1e-5f);
  #pragma unroll
  for (int i = 0; i < 4; ++i) {
    int c = tid + i * 256;
    out[(size_t)row * C + c] = f2bf((v[i] - mean) * rstd * g[c] + b[c]);
  }
}

// ------- bf16 WMMA GEMM: out = A[M,K] @ Wt[N,K]^T + bias --------------------
// Double-buffered LDS filled by async global->LDS copies (ASYNCcnt).
// MODE 0: bf16 out.  MODE 1: fp32 out = resid + acc.  MODE 2: bf16 gelu(acc).
template <int MODE>
__global__ __launch_bounds__(256)
void k_gemm(const us* __restrict__ A, const us* __restrict__ Wt,
            const float* __restrict__ bias, const float* __restrict__ resid,
            void* __restrict__ outp, int M, int N, int K) {
  __shared__ __align__(16) us lA[2][128 * 32];
  __shared__ __align__(16) us lW[2][64 * 32];
  const int tid  = threadIdx.x;
  const int lane = tid & 31, wid = tid >> 5;
  const int wm = wid & 3, wn = wid >> 2;
  const int l15 = lane & 15, khalf = (lane >> 4) << 3;
  const int mbase = blockIdx.y * 128, nbase = blockIdx.x * 64;

  const int arow0 = tid >> 2, acc8 = (tid & 3) << 3;   // A: 2 x 16B / thread
  const int wrow  = tid >> 2, wcc8 = (tid & 3) << 3;   // W: 1 x 16B / thread

  auto issue = [&](int p, int k0) {
    async_copy16(&lA[p][arow0 * 32 + acc8],
                 A  + (size_t)(mbase + arow0)      * K + k0 + acc8);
    async_copy16(&lA[p][(arow0 + 64) * 32 + acc8],
                 A  + (size_t)(mbase + arow0 + 64) * K + k0 + acc8);
    async_copy16(&lW[p][wrow * 32 + wcc8],
                 Wt + (size_t)(nbase + wrow)       * K + k0 + wcc8);
  };

  v8f c[2][2];
  c[0][0] = zero8(); c[0][1] = zero8(); c[1][0] = zero8(); c[1][1] = zero8();

  const int nk = K >> 5;
  issue(0, 0);
  wait_async0();
  __syncthreads();
  for (int j = 0; j < nk; ++j) {
    const int p = j & 1;
    if (j + 1 < nk) issue((j + 1) & 1, (j + 1) << 5);
    FragB16 a[2], bfr[2];
    #pragma unroll
    for (int im = 0; im < 2; ++im) {
      int r = wm * 32 + im * 16 + l15;
      a[im].q[0] = *(const uint4*)&lA[p][r * 32 + khalf];
      a[im].q[1] = *(const uint4*)&lA[p][r * 32 + 16 + khalf];
    }
    #pragma unroll
    for (int in = 0; in < 2; ++in) {
      int nc = wn * 32 + in * 16 + l15;
      bfr[in].q[0] = *(const uint4*)&lW[p][nc * 32 + khalf];
      bfr[in].q[1] = *(const uint4*)&lW[p][nc * 32 + 16 + khalf];
    }
    #pragma unroll
    for (int im = 0; im < 2; ++im)
      #pragma unroll
      for (int in = 0; in < 2; ++in)
        c[im][in] = wmma_bf16(a[im], bfr[in], c[im][in]);
    if (j + 1 < nk) wait_async0();
    __syncthreads();
  }

  const int hi8 = (lane >> 4) << 3;
  #pragma unroll
  for (int im = 0; im < 2; ++im)
    #pragma unroll
    for (int in = 0; in < 2; ++in)
      #pragma unroll
      for (int r = 0; r < 8; ++r) {
        int row = mbase + wm * 32 + im * 16 + r + hi8;
        int col = nbase + wn * 32 + in * 16 + l15;
        size_t idx = (size_t)row * N + col;
        float v = c[im][in][r] + bias[col];
        if (MODE == 0) {
          ((us*)outp)[idx] = f2bf(v);
        } else if (MODE == 1) {
          ((float*)outp)[idx] = resid[idx] + v;
        } else {
          float gv = 0.5f * v * (1.0f + erff(v * 0.70710678118654752f));
          ((us*)outp)[idx] = f2bf(gv);
        }
      }
}

// ---------------- Flash attention: one wave per (b, h, 16-row q-tile) -------
// V pre-transposed to Vt[(b*H+h)*64 + d][T].  Row-sum folded into a WMMA with
// an all-ones B operand; row-max via DPP16 (no LDS permutes).
__global__ __launch_bounds__(256)
void k_flash(const us* __restrict__ Qb, int q_rs, int q_col0,
             const us* __restrict__ Kb, int k_rs, int k_col0,
             const us* __restrict__ Vt,
             us* __restrict__ Ob, int o_rs,
             int T, int H, int causal, float scale) {
  __shared__ __align__(16) us lP[8][16 * 32];
  const int lane = threadIdx.x & 31, wid = threadIdx.x >> 5;
  const int gw = blockIdx.x * 8 + wid;
  const int nQT = T >> 4;
  const int qt = gw % nQT;
  const int h  = (gw / nQT) % H;
  const int b  = gw / (nQT * H);
  const int l15 = lane & 15, khalf = (lane >> 4) << 3, hi8 = khalf;
  const int qcol = q_col0 + h * 64, kcol = k_col0 + h * 64;
  const size_t brow = (size_t)b * T;
  const us* vt = Vt + ((size_t)(b * H + h) * 64) * T;

  FragB16 ones;
  {
    uint4 o; o.x = o.y = o.z = o.w = 0x3F803F80u;   // bf16 1.0 pairs
    ones.q[0] = o; ones.q[1] = o;
  }
  FragB16 aq[2];
  {
    const us* qr = Qb + (brow + qt * 16 + l15) * q_rs + qcol;
    #pragma unroll
    for (int kc = 0; kc < 2; ++kc) {
      aq[kc].q[0] = *(const uint4*)(qr + kc * 32 + khalf);
      aq[kc].q[1] = *(const uint4*)(qr + kc * 32 + 16 + khalf);
    }
  }
  float mrow[8], lrow[8], alph[8];
  v8f oc[4];
  #pragma unroll
  for (int r = 0; r < 8; ++r) { mrow[r] = -__builtin_inff(); lrow[r] = 0.f; }
  oc[0] = zero8(); oc[1] = zero8(); oc[2] = zero8(); oc[3] = zero8();

  const int rbase = qt * 16 + hi8;
  const int nsteps = causal ? (qt / 2 + 1) : (T >> 5);

  for (int j = 0; j < nsteps; ++j) {
    v8f s[2];
    #pragma unroll
    for (int ns = 0; ns < 2; ++ns) {
      const us* kr = Kb + (brow + j * 32 + ns * 16 + l15) * k_rs + kcol;
      FragB16 bk0, bk1;
      bk0.q[0] = *(const uint4*)(kr + khalf);
      bk0.q[1] = *(const uint4*)(kr + 16 + khalf);
      bk1.q[0] = *(const uint4*)(kr + 32 + khalf);
      bk1.q[1] = *(const uint4*)(kr + 48 + khalf);
      v8f acc = zero8();
      acc = wmma_bf16(aq[0], bk0, acc);
      acc = wmma_bf16(aq[1], bk1, acc);
      s[ns] = acc;
    }
    const int col0 = j * 32 + l15;
    #pragma unroll
    for (int r = 0; r < 8; ++r) {
      float s0 = s[0][r] * scale;
      float s1 = s[1][r] * scale;
      const int rowg = rbase + r;
      if (causal) {
        if (col0 > rowg)      s0 = -__builtin_inff();
        if (col0 + 16 > rowg) s1 = -__builtin_inff();
      }
      float t = dppmax16(fmaxf(s0, s1));
      float mn = fmaxf(mrow[r], t);
      float a = __expf(mrow[r] - mn);
      alph[r] = a;
      mrow[r] = mn;
      #pragma unroll
      for (int d = 0; d < 4; ++d) oc[d][r] *= a;
      lP[wid][(r + hi8) * 32 + l15]      = f2bf(__expf(s0 - mn));
      lP[wid][(r + hi8) * 32 + 16 + l15] = f2bf(__expf(s1 - mn));
    }
    asm volatile("s_wait_dscnt 0" ::: "memory");
    FragB16 ap;
    ap.q[0] = *(const uint4*)&lP[wid][l15 * 32 + khalf];
    ap.q[1] = *(const uint4*)&lP[wid][l15 * 32 + 16 + khalf];
    {
      v8f sums = wmma_bf16(ap, ones, zero8());
      #pragma unroll
      for (int r = 0; r < 8; ++r) lrow[r] = lrow[r] * alph[r] + sums[r];
    }
    #pragma unroll
    for (int d = 0; d < 4; ++d) {
      const us* vr = vt + (size_t)(d * 16 + l15) * T + j * 32;
      FragB16 bv;
      bv.q[0] = *(const uint4*)(vr + khalf);
      bv.q[1] = *(const uint4*)(vr + 16 + khalf);
      oc[d] = wmma_bf16(ap, bv, oc[d]);
    }
  }
  #pragma unroll
  for (int r = 0; r < 8; ++r) {
    float inv = 1.0f / lrow[r];
    const int row = rbase + r;
    #pragma unroll
    for (int d = 0; d < 4; ++d)
      Ob[(brow + row) * o_rs + h * 64 + d * 16 + l15] = f2bf(oc[d][r] * inv);
  }
}

// ----------------------------- host launcher --------------------------------
extern "C" void kernel_launch(void* const* d_in, const int* in_sizes, int n_in,
                              void* d_out, int out_size, void* d_ws,
                              size_t ws_size, hipStream_t stream) {
  (void)in_sizes; (void)n_in; (void)out_size; (void)ws_size;
  constexpr int B = 4, T = 1024, C = 1024, H = 16, F = 4096;
  constexpr int BT = B * T;
  const float* x     = (const float*)d_in[0];
  const float* ctx   = (const float*)d_in[1];
  const float* ln1_g = (const float*)d_in[2];
  const float* ln1_b = (const float*)d_in[3];
  const float* qkv_w = (const float*)d_in[4];
  const float* qkv_b = (const float*)d_in[5];
  const float* aow   = (const float*)d_in[6];
  const float* aob   = (const float*)d_in[7];
  const float* lnc_g = (const float*)d_in[8];
  const float* lnc_b = (const float*)d_in[9];
  const float* q_w   = (const float*)d_in[10];
  const float* q_b   = (const float*)d_in[11];
  const float* kv_w  = (const float*)d_in[12];
  const float* kv_b  = (const float*)d_in[13];
  const float* cow   = (const float*)d_in[14];
  const float* cob   = (const float*)d_in[15];
  const float* ln2_g = (const float*)d_in[16];
  const float* ln2_b = (const float*)d_in[17];
  const float* w1    = (const float*)d_in[18];
  const float* b1    = (const float*)d_in[19];
  const float* w2    = (const float*)d_in[20];
  const float* b2    = (const float*)d_in[21];

  char* p = (char*)d_ws;
  auto carve = [&](size_t bytes) -> char* {
    char* r = p; p += (bytes + 255) & ~(size_t)255; return r;
  };
  us* wqkv = (us*)carve((size_t)C * 3 * C * 2);   // [3C][C] transposed
  us* wao  = (us*)carve((size_t)C * C * 2);
  us* wq   = (us*)carve((size_t)C * C * 2);
  us* wkv  = (us*)carve((size_t)C * 2 * C * 2);
  us* wco  = (us*)carve((size_t)C * C * 2);
  us* w1b  = (us*)carve((size_t)C * F * 2);
  us* w2b  = (us*)carve((size_t)F * C * 2);
  us* hbuf = (us*)carve((size_t)BT * C * 2);
  us* qkvb = (us*)carve((size_t)BT * 3 * C * 2);
  us* obuf = (us*)carve((size_t)BT * C * 2);
  us* ctxb = (us*)carve((size_t)BT * C * 2);
  us* vtb  = (us*)carve((size_t)BT * C * 2);      // Vt[(b*H+h)*64 + d][T]
  us* f1b  = (us*)carve((size_t)BT * F * 2);
  float* resid = (float*)carve((size_t)BT * C * 4);
  us* qxb = qkvb;                    // reuse after self-attention
  us* kvb = qkvb + (size_t)BT * C;

  auto cvtT = [&](const float* s, us* d, int K, int N) {
    k_cvt_t<<<dim3(N / 32, K / 32), 256, 0, stream>>>(s, d, K, N);
  };
  cvtT(qkv_w, wqkv, C, 3 * C);
  cvtT(aow,   wao,  C, C);
  cvtT(q_w,   wq,   C, C);
  cvtT(kv_w,  wkv,  C, 2 * C);
  cvtT(cow,   wco,  C, C);
  cvtT(w1,    w1b,  C, F);
  cvtT(w2,    w2b,  F, C);
  {
    int n4 = BT * C / 4;
    k_cvt_bf16<<<(n4 + 255) / 256, 256, 0, stream>>>(ctx, ctxb, n4);
  }

  const int flashBlocks = (B * H * (T / 16)) / 8;   // 512
  const float scale = 0.125f;                       // 1/sqrt(64)

  // 1) self-attention branch
  k_layernorm<<<BT, 256, 0, stream>>>(x, ln1_g, ln1_b, hbuf, C);
  k_gemm<0><<<dim3(3 * C / 64, BT / 128), 256, 0, stream>>>(
      hbuf, wqkv, qkv_b, nullptr, qkvb, BT, 3 * C, C);
  k_vt<<<dim3(T / 32, B * H), 256, 0, stream>>>(qkvb, 3 * C, 2 * C, vtb, T, H);
  k_flash<<<flashBlocks, 256, 0, stream>>>(
      qkvb, 3 * C, 0, qkvb, 3 * C, C, vtb, obuf, C, T, H, 1, scale);
  k_gemm<1><<<dim3(C / 64, BT / 128), 256, 0, stream>>>(
      obuf, wao, aob, x, resid, BT, C, C);

  // 2) cross-attention branch
  k_layernorm<<<BT, 256, 0, stream>>>(resid, lnc_g, lnc_b, hbuf, C);
  k_gemm<0><<<dim3(C / 64, BT / 128), 256, 0, stream>>>(
      hbuf, wq, q_b, nullptr, qxb, BT, C, C);
  k_gemm<0><<<dim3(2 * C / 64, BT / 128), 256, 0, stream>>>(
      ctxb, wkv, kv_b, nullptr, kvb, BT, 2 * C, C);
  k_vt<<<dim3(T / 32, B * H), 256, 0, stream>>>(kvb, 2 * C, C, vtb, T, H);
  k_flash<<<flashBlocks, 256, 0, stream>>>(
      qxb, C, 0, kvb, 2 * C, 0, vtb, obuf, C, T, H, 0, scale);
  k_gemm<1><<<dim3(C / 64, BT / 128), 256, 0, stream>>>(
      obuf, wco, cob, resid, resid, BT, C, C);

  // 3) FFN
  k_layernorm<<<BT, 256, 0, stream>>>(resid, ln2_g, ln2_b, hbuf, C);
  k_gemm<2><<<dim3(F / 64, BT / 128), 256, 0, stream>>>(
      hbuf, w1b, b1, nullptr, f1b, BT, F, C);
  k_gemm<1><<<dim3(C / 64, BT / 128), 256, 0, stream>>>(
      f1b, w2b, b2, resid, (float*)d_out, BT, C, F);
}
